// AttentionTensorCast_84447646973961
// MI455X (gfx1250) — 8-bit matrix operand order repaired and checked by reading, compiled, not run on hardware
//
#include <hip/hip_runtime.h>
#include <hip/hip_bf16.h>
#include <stdint.h>

#define BQ 4
#define HQ 32
#define KVHN 8
#define DD 128
#define LQN 256
#define SSN 4096
#define BLKN 16
#define BPSN (SSN / BLKN)
#define NUMBLK (BQ * BPSN)
#define TTN (BQ * LQN)
#define GGN (HQ / KVHN)
#define SM_SCALE 0.08838834764831845f

typedef int v8i __attribute__((ext_vector_type(8)));

#if __has_builtin(__builtin_amdgcn_tensor_load_to_lds) && __has_builtin(__builtin_amdgcn_s_wait_tensorcnt)
#define USE_TDM 1
#else
#define USE_TDM 0
#endif

__device__ __forceinline__ const int8_t* kv_row_ptr(const int8_t* cache, const int* bt,
                                                    const int8_t* newkv, int plane,
                                                    int b, int kvh, int s) {
  if (s >= SSN - LQN) {
    return newkv + (((size_t)(b * KVHN + kvh) * LQN + (s - (SSN - LQN))) * DD);
  }
  int blk = bt[b * BPSN + (s >> 4)];
  return cache + ((((size_t)plane * NUMBLK + blk) * BLKN + (s & 15)) * KVHN + kvh) * DD;
}

__device__ __forceinline__ int quant8(float x, float sc, float off) {
  float r = rintf(x / sc + off);
  r = fminf(fmaxf(r, -128.f), 127.f);
  return (int)r;
}

__device__ __forceinline__ int sumbytes(int v) {
  return (int)(char)(v) + (int)(char)(v >> 8) + (int)(char)(v >> 16) + (v >> 24);
}

#if USE_TDM
__device__ __forceinline__ void tdm_load_tile(uint32_t lds_addr, const int8_t* gptr,
                                              int rows, int stride_bytes) {
  typedef unsigned int v4u __attribute__((ext_vector_type(4)));
  typedef int v4s __attribute__((ext_vector_type(4)));
  uint64_t ga = (uint64_t)(uintptr_t)gptr;
  v4u g0;
  g0[0] = 1u;
  g0[1] = lds_addr;
  g0[2] = (uint32_t)ga;
  g0[3] = (uint32_t)((ga >> 32) & 0x01FFFFFFu) | (2u << 30);
  v8i g1;
  g1[0] = (1 << 20) | (4 << 22) | (3 << 25);
  g1[1] = (DD & 0xffff) << 16;
  g1[2] = (rows & 0xffff) << 16;
  g1[3] = (DD & 0xffff) << 16;
  g1[4] = (rows & 0xffff);
  g1[5] = stride_bytes;
  g1[6] = 0;
  g1[7] = 0;
  v4s gz = {0, 0, 0, 0};
#if __has_include(<hip/amd_detail/amd_gfx1250_TDM.h>)
  v8i gz8 = {0, 0, 0, 0, 0, 0, 0, 0};
  __builtin_amdgcn_tensor_load_to_lds(g0, g1, gz, gz, gz8, 0);
#else
  __builtin_amdgcn_tensor_load_to_lds(g0, g1, gz, gz, 0);
#endif
}
#endif

__global__ void prep_q_kernel(const float* __restrict__ query,
                              const float* qsp, const float* qop,
                              int8_t* __restrict__ Qq, int* __restrict__ qsum) {
  const float qs = qsp[0], qo = qop[0];
  int idx = blockIdx.x * blockDim.x + threadIdx.x;
  int row = idx >> 2, part = idx & 3;
  const float* src = query + (size_t)row * DD + part * 32;
  int8_t* dst = Qq + (size_t)row * DD + part * 32;
  int sum = 0;
#pragma unroll
  for (int g = 0; g < 8; ++g) {
    int w = 0;
#pragma unroll
    for (int e = 0; e < 4; ++e) {
      int q = quant8(src[g * 4 + e], qs, qo);
      sum += q;
      w |= (q & 0xff) << (8 * e);
    }
    ((int*)dst)[g] = w;
  }
  sum += __shfl_xor(sum, 1, 32);
  sum += __shfl_xor(sum, 2, 32);
  if (part == 0) qsum[row] = sum;
}

__global__ void prep_kv_kernel(const float* __restrict__ keyf, const float* __restrict__ valf,
                               const float* ksp, const float* kop,
                               int8_t* __restrict__ newK, int8_t* __restrict__ newV) {
  const float ks = ksp[0], ko = kop[0];
  int idx = blockIdx.x * blockDim.x + threadIdx.x;
  int row = idx >> 2, part = idx & 3;
  int kvh = row % KVHN;
  int t = row / KVHN;
  int b = t / LQN, lq = t % LQN;
  const float* sk = keyf + ((size_t)t * KVHN + kvh) * DD + part * 32;
  const float* sv = valf + ((size_t)t * KVHN + kvh) * DD + part * 32;
  size_t dofs = ((size_t)(b * KVHN + kvh) * LQN + lq) * DD + part * 32;
  int8_t* dk = newK + dofs;
  int8_t* dv = newV + dofs;
#pragma unroll
  for (int g = 0; g < 8; ++g) {
    int wk = 0, wv = 0;
#pragma unroll
    for (int e = 0; e < 4; ++e) {
      wk |= (quant8(sk[g * 4 + e], ks, ko) & 0xff) << (8 * e);
      wv |= (quant8(sv[g * 4 + e], ks, ko) & 0xff) << (8 * e);
    }
    ((int*)dk)[g] = wk;
    ((int*)dv)[g] = wv;
  }
}

__global__ void ksum_kernel(const int8_t* __restrict__ cache, const int* __restrict__ bt,
                            const int8_t* __restrict__ newK, int* __restrict__ ksum) {
  int idx = blockIdx.x * blockDim.x + threadIdx.x;
  int row = idx >> 2, part = idx & 3;
  int s = row % SSN;
  int kvh = (row / SSN) % KVHN;
  int b = row / (SSN * KVHN);
  const int8_t* p = kv_row_ptr(cache, bt, newK, 0, b, kvh, s) + part * 32;
  int sum = 0;
#pragma unroll
  for (int g = 0; g < 8; ++g) sum += sumbytes(((const int*)p)[g]);
  sum += __shfl_xor(sum, 1, 32);
  sum += __shfl_xor(sum, 2, 32);
  if (part == 0) ksum[row] = sum;
}

__global__ void vsum_kernel(const int8_t* __restrict__ cache, const int* __restrict__ bt,
                            const int8_t* __restrict__ newV, int* __restrict__ vsum) {
  __shared__ int acc[DD];
  int b = blockIdx.x / KVHN, kvh = blockIdx.x % KVHN;
  int tid = threadIdx.x;
  if (tid < DD) acc[tid] = 0;
  __syncthreads();
  int d4 = (tid & 31) * 4;
  int sr = tid >> 5;
  int p0 = 0, p1 = 0, p2 = 0, p3 = 0;
  for (int s = sr * 512; s < sr * 512 + 512; ++s) {
    const int8_t* vp = kv_row_ptr(cache, bt, newV, 1, b, kvh, s);
    int w = *(const int*)(vp + d4);
    p0 += (int)(char)w; p1 += (int)(char)(w >> 8);
    p2 += (int)(char)(w >> 16); p3 += (w >> 24);
  }
  atomicAdd(&acc[d4 + 0], p0);
  atomicAdd(&acc[d4 + 1], p1);
  atomicAdd(&acc[d4 + 2], p2);
  atomicAdd(&acc[d4 + 3], p3);
  __syncthreads();
  if (tid < DD) vsum[blockIdx.x * DD + tid] = acc[tid];
}

__global__ __launch_bounds__(256) void attn_kernel(
    const int8_t* __restrict__ Qq, const int* __restrict__ qsum,
    const int8_t* __restrict__ newK, const int8_t* __restrict__ newV,
    const int* __restrict__ ksum, const int* __restrict__ vsum,
    const int8_t* __restrict__ cache, const int* __restrict__ bt,
    const int* __restrict__ seq_lens,
    const float* qsp, const float* qop, const float* ksp, const float* kop,
    const float* psp, const float* pop,
    float* __restrict__ out) {

  __shared__ int8_t Klds[64 * 144];
  __shared__ int8_t Vrow[64 * 144];
  __shared__ int8_t Vtr[128 * 80];
  __shared__ int8_t Pbuf[8][16 * 72];

  const float qs = qsp[0], qo = qop[0], ks = ksp[0], ko = kop[0], ps = psp[0], po = pop[0];

  const int tid = threadIdx.x;
  const int lane = tid & 31;
  const int wave = tid >> 5;
  const int wg = blockIdx.x;
  const int qt = wg & 7;
  const int kvh = (wg >> 3) & 7;
  const int b = wg >> 6;

  const int Rbase = qt * 128 + wave * 16;
  const int g = Rbase / LQN;
  const int lq0 = Rbase % LQN;
  const int h = kvh * GGN + g;
  const int seqlen = seq_lens[b];

  const int mrow = lane & 15;
  const int hi = lane >> 4;
  const int colk = mrow;

  const int pat[8] = {0, 4, 8, 12, 32, 36, 40, 44};

#if USE_TDM
  const uint32_t kldsbase = (uint32_t)(uintptr_t)(&Klds[0]);
  const uint32_t vldsbase = (uint32_t)(uintptr_t)(&Vrow[0]);
#endif

  const int8_t* qrow = Qq + (((size_t)(b * LQN + lq0 + mrow)) * HQ + h) * DD;
  __builtin_prefetch(qrow, 0, 3);
  v8i a0, a1;
#pragma unroll
  for (int j = 0; j < 8; ++j) {
    int off = pat[j] + hi * 16;
    a0[j] = *(const int*)(qrow + off);
    a1[j] = *(const int*)(qrow + 64 + off);
  }

  const float Ascale = qs * ks * SM_SCALE;
  float Bi[8];
  int qpos[8];
#pragma unroll
  for (int i = 0; i < 8; ++i) {
    int r = i + hi * 8;
    int rq = qsum[(size_t)(b * LQN + lq0 + r) * HQ + h];
    Bi[i] = Ascale * (-ko * (float)rq + (float)DD * qo * ko);
    qpos[i] = seqlen - LQN + (lq0 + r);
  }

  float mi[8], li[8];
#pragma unroll
  for (int i = 0; i < 8; ++i) { mi[i] = -3.0e38f; li[i] = 0.f; }

  const int ksbase = (b * KVHN + kvh) * SSN;

  for (int chunk = 0; chunk < SSN / 64; ++chunk) {
    __syncthreads();
#if USE_TDM
    if (wave == 0) {
      int s0 = chunk * 64;
      if (s0 >= SSN - LQN) {
        const int8_t* src = newK + ((size_t)(b * KVHN + kvh) * LQN + (s0 - (SSN - LQN))) * DD;
        tdm_load_tile(kldsbase, src, 64, DD);
      } else {
#pragma unroll
        for (int p = 0; p < 4; ++p) {
          int blk = bt[b * BPSN + ((s0 + p * 16) >> 4)];
          const int8_t* src = cache + (((size_t)blk * BLKN * KVHN) + kvh) * DD;
          tdm_load_tile(kldsbase + p * 16 * 144, src, 16, KVHN * DD);
        }
      }
      __builtin_amdgcn_s_wait_tensorcnt(0);
    }
#else
    {
      int key = tid >> 2, part = tid & 3;
      int s = chunk * 64 + key;
      const int8_t* src = kv_row_ptr(cache, bt, newK, 0, b, kvh, s);
      *(int4*)(Klds + key * 144 + part * 16) = *(const int4*)(src + part * 16);
      *(int4*)(Klds + key * 144 + 64 + part * 16) = *(const int4*)(src + 64 + part * 16);
    }
#endif
    __syncthreads();
#pragma unroll
    for (int st = 0; st < 4; ++st) {
      const int8_t* kp = Klds + (st * 16 + colk) * 144 + hi * 16;
      int4 x0 = *(const int4*)(kp + 0);
      int4 x1 = *(const int4*)(kp + 32);
      int4 y0 = *(const int4*)(kp + 64);
      int4 y1 = *(const int4*)(kp + 96);
      v8i bk0 = {x0.x, x0.y, x0.z, x0.w, x1.x, x1.y, x1.z, x1.w};
      v8i bk1 = {y0.x, y0.y, y0.z, y0.w, y1.x, y1.y, y1.z, y1.w};
      v8i acc = {0, 0, 0, 0, 0, 0, 0, 0};
      acc = __builtin_amdgcn_wmma_i32_16x16x64_iu8(true, a0, true, bk0, acc, false, false);
      acc = __builtin_amdgcn_wmma_i32_16x16x64_iu8(true, a1, true, bk1, acc, false, false);
      int scol = chunk * 64 + st * 16 + colk;
      float Ccol = Ascale * (-qo * (float)ksum[ksbase + scol]);
#pragma unroll
      for (int i = 0; i < 8; ++i) {
        float sv = (float)acc[i] * Ascale + Bi[i] + Ccol;
        sv = (scol <= qpos[i]) ? sv : -1e9f;
        float mo = mi[i];
        float mn = fmaxf(mo, sv);
        li[i] = li[i] * __expf(mo - mn) + __expf(sv - mn);
        mi[i] = mn;
      }
    }
  }

  float Mrow[8], Linv[8];
#pragma unroll
  for (int i = 0; i < 8; ++i) {
    float m = mi[i], l = li[i];
#pragma unroll
    for (int step = 1; step < 16; step <<= 1) {
      float mo = __shfl_xor(m, step, 32);
      float lo = __shfl_xor(l, step, 32);
      float mn = fmaxf(m, mo);
      l = l * __expf(m - mn) + lo * __expf(mo - mn);
      m = mn;
    }
    Mrow[i] = m;
    Linv[i] = 1.0f / l;
  }

  v8i oacc[8];
  v8i vzero = {0, 0, 0, 0, 0, 0, 0, 0};
#pragma unroll
  for (int n = 0; n < 8; ++n) oacc[n] = vzero;
  int psum[8];
#pragma unroll
  for (int i = 0; i < 8; ++i) psum[i] = 0;
  const float inv_ps = 1.0f / ps;

  for (int chunk = 0; chunk < SSN / 64; ++chunk) {
    __syncthreads();
#if USE_TDM
    if (wave == 0) {
      int s0 = chunk * 64;
      if (s0 >= SSN - LQN) {
        size_t nofs = ((size_t)(b * KVHN + kvh) * LQN + (s0 - (SSN - LQN))) * DD;
        tdm_load_tile(kldsbase, newK + nofs, 64, DD);
        tdm_load_tile(vldsbase, newV + nofs, 64, DD);
      } else {
#pragma unroll
        for (int p = 0; p < 4; ++p) {
          int blk = bt[b * BPSN + ((s0 + p * 16) >> 4)];
          size_t kofs = (((size_t)blk * BLKN * KVHN) + kvh) * DD;
          size_t vofs = ((((size_t)NUMBLK + blk) * BLKN * KVHN) + kvh) * DD;
          tdm_load_tile(kldsbase + p * 16 * 144, cache + kofs, 16, KVHN * DD);
          tdm_load_tile(vldsbase + p * 16 * 144, cache + vofs, 16, KVHN * DD);
        }
      }
      __builtin_amdgcn_s_wait_tensorcnt(0);
    }
#else
    {
      int key = tid >> 2, part = tid & 3;
      int s = chunk * 64 + key;
      const int8_t* srck = kv_row_ptr(cache, bt, newK, 0, b, kvh, s);
      const int8_t* srcv = kv_row_ptr(cache, bt, newV, 1, b, kvh, s);
      *(int4*)(Klds + key * 144 + part * 16) = *(const int4*)(srck + part * 16);
      *(int4*)(Klds + key * 144 + 64 + part * 16) = *(const int4*)(srck + 64 + part * 16);
      *(int4*)(Vrow + key * 144 + part * 16) = *(const int4*)(srcv + part * 16);
      *(int4*)(Vrow + key * 144 + 64 + part * 16) = *(const int4*)(srcv + 64 + part * 16);
    }
#endif
    __syncthreads();
#pragma unroll
    for (int j = 0; j < 8; ++j) {
      int idx = tid + j * 256;
      int d = idx >> 4;
      int k4 = (idx & 15) * 4;
      int w = (int)(uint8_t)Vrow[(k4 + 0) * 144 + d]
            | ((int)(uint8_t)Vrow[(k4 + 1) * 144 + d] << 8)
            | ((int)(uint8_t)Vrow[(k4 + 2) * 144 + d] << 16)
            | ((int)(uint8_t)Vrow[(k4 + 3) * 144 + d] << 24);
      *(int*)(Vtr + d * 80 + k4) = w;
    }
    __syncthreads();

#pragma unroll
    for (int st = 0; st < 4; ++st) {
      const int8_t* kp = Klds + (st * 16 + colk) * 144 + hi * 16;
      int4 x0 = *(const int4*)(kp + 0);
      int4 x1 = *(const int4*)(kp + 32);
      int4 y0 = *(const int4*)(kp + 64);
      int4 y1 = *(const int4*)(kp + 96);
      v8i bk0 = {x0.x, x0.y, x0.z, x0.w, x1.x, x1.y, x1.z, x1.w};
      v8i bk1 = {y0.x, y0.y, y0.z, y0.w, y1.x, y1.y, y1.z, y1.w};
      v8i acc = {0, 0, 0, 0, 0, 0, 0, 0};
      acc = __builtin_amdgcn_wmma_i32_16x16x64_iu8(true, a0, true, bk0, acc, false, false);
      acc = __builtin_amdgcn_wmma_i32_16x16x64_iu8(true, a1, true, bk1, acc, false, false);
      int scol = chunk * 64 + st * 16 + colk;
      float Ccol = Ascale * (-qo * (float)ksum[ksbase + scol]);
#pragma unroll
      for (int i = 0; i < 8; ++i) {
        float sv = (float)acc[i] * Ascale + Bi[i] + Ccol;
        sv = (scol <= qpos[i]) ? sv : -1e9f;
        float p = __expf(sv - Mrow[i]) * Linv[i];
        float r = rintf(p * inv_ps + po);
        r = fminf(fmaxf(r, -128.f), 127.f);
        int pq = (int)r;
        psum[i] += pq;
        Pbuf[wave][(i + hi * 8) * 72 + st * 16 + colk] = (int8_t)pq;
      }
    }
    __builtin_amdgcn_wave_barrier();

    v8i ap;
    const int8_t* pr = &Pbuf[wave][mrow * 72];
#pragma unroll
    for (int j = 0; j < 8; ++j) ap[j] = *(const int*)(pr + pat[j] + hi * 16);

#pragma unroll
    for (int n = 0; n < 8; ++n) {
      const int8_t* vp = Vtr + (n * 16 + colk) * 80 + hi * 16;
      int4 z0 = *(const int4*)(vp + 0);
      int4 z1 = *(const int4*)(vp + 32);
      v8i bv = {z0.x, z0.y, z0.z, z0.w, z1.x, z1.y, z1.z, z1.w};
      oacc[n] = __builtin_amdgcn_wmma_i32_16x16x64_iu8(true, ap, true, bv, oacc[n], false, false);
    }
  }

#pragma unroll
  for (int i = 0; i < 8; ++i) {
    int v = psum[i];
    v += __shfl_xor(v, 1, 32);
    v += __shfl_xor(v, 2, 32);
    v += __shfl_xor(v, 4, 32);
    v += __shfl_xor(v, 8, 32);
    psum[i] = v;
  }

  const float oscale = ps * ks;
  const float cterm = (float)SSN * po * ko;
#pragma unroll
  for (int n = 0; n < 8; ++n) {
    int d = n * 16 + colk;
    float vsn = (float)vsum[(b * KVHN + kvh) * DD + d];
#pragma unroll
    for (int i = 0; i < 8; ++i) {
      int r = i + hi * 8;
      size_t t = (size_t)(b * LQN + lq0 + r);
      float o = oscale * ((float)oacc[n][i] - ko * (float)psum[i] - po * vsn + cterm);
      out[(t * HQ + h) * DD + d] = o;
    }
  }
}

extern "C" void kernel_launch(void* const* d_in, const int* in_sizes, int n_in,
                              void* d_out, int out_size, void* d_ws, size_t ws_size,
                              hipStream_t stream) {
  const float* query = (const float*)d_in[0];
  const float* keyf = (const float*)d_in[1];
  const float* valf = (const float*)d_in[2];
  const int8_t* cache = (const int8_t*)d_in[3];
  const int* bt = (const int*)d_in[4];
  const int* seq_lens = (const int*)d_in[7];
  const float* qsp = (const float*)d_in[9];
  const float* qop = (const float*)d_in[10];
  const float* ksp = (const float*)d_in[11];
  const float* kop = (const float*)d_in[12];
  const float* psp = (const float*)d_in[13];
  const float* pop = (const float*)d_in[14];
  float* out = (float*)d_out;

  uint8_t* w = (uint8_t*)d_ws;
  int8_t* Qq = (int8_t*)w;   w += (size_t)TTN * HQ * DD;
  int* qsum = (int*)w;       w += (size_t)TTN * HQ * 4;
  int8_t* newK = (int8_t*)w; w += (size_t)BQ * KVHN * LQN * DD;
  int8_t* newV = (int8_t*)w; w += (size_t)BQ * KVHN * LQN * DD;
  int* ksum = (int*)w;       w += (size_t)BQ * KVHN * SSN * 4;
  int* vsum = (int*)w;       w += (size_t)BQ * KVHN * DD * 4;

  prep_q_kernel<<<(TTN * HQ * 4) / 256, 256, 0, stream>>>(query, qsp, qop, Qq, qsum);
  prep_kv_kernel<<<(TTN * KVHN * 4) / 256, 256, 0, stream>>>(keyf, valf, ksp, kop, newK, newV);
  ksum_kernel<<<(BQ * KVHN * SSN * 4) / 256, 256, 0, stream>>>(cache, bt, newK, ksum);
  vsum_kernel<<<BQ * KVHN, 256, 0, stream>>>(cache, bt, newV, vsum);
  attn_kernel<<<BQ * KVHN * 8, 256, 0, stream>>>(Qq, qsum, newK, newV, ksum, vsum,
                                                 cache, bt, seq_lens,
                                                 qsp, qop, ksp, kop, psp, pop, out);
}
